// EncoderDecoderWrapper_21852793602567
// MI455X (gfx1250) — compile-verified, matrix operand order checked by reading, not run on hardware
//
#include <hip/hip_runtime.h>
#include <hip/hip_bf16.h>
#include <math.h>

// ---------------------------------------------------------------------------
// Fused encoder/decoder LSTM+attention for MI455X (gfx1250, wave32, WMMA).
//
// B=2048 S=168 F=7 H=128 P=96.  One workgroup owns BT=32 batch rows and runs
// the full 168-step encoder + 96-step decoder recurrence with no inter-block
// sync.  All GEMMs use v_wmma_f32_16x16x32_bf16 (f32 accumulate).  Weights are
// pre-converted to padded bf16 in d_ws by an init kernel.  Recurrent weights
// (W_ih_enc|W_hh_enc, later W_hh_dec) are staged into LDS with
// GLOBAL_LOAD_ASYNC_TO_LDS_B128 so the 264 serial steps read B-fragments from
// LDS instead of L2.  286KB LDS -> exactly 1 workgroup per 320KB WGP: all 64
// blocks land on distinct WGPs.  encoder_output is stored bf16 in d_ws
// (88 MB -> L2-resident across the 96 decoder re-reads).  Activations use raw
// v_rcp_f32/v_exp_f32; softmax is wave-local via __shfl_xor.
// ---------------------------------------------------------------------------

typedef __bf16 bf16_t;
typedef __attribute__((ext_vector_type(16))) __bf16 v16bf;
typedef __attribute__((ext_vector_type(8)))  float  v8f;

#define B_  2048
#define S_  168
#define F_  7
#define H_  128
#define P_  96
#define G_  512          // 4*H
#define BT  32           // batch rows per workgroup
#define NT_THREADS 256   // 8 wave32s

// d_ws byte layout (all 256B aligned); WIHX and WHH are contiguous on purpose
// (single 160KB async bulk copy into LDS).
#define WS_ENC   0u                                   // bf16 [B][S][H]      88,080,384 B
#define WS_WIHX  88080384u                            // bf16 [512][32]  (W_ih_enc padded 7->32)
#define WS_WHH   (WS_WIHX + 32768u)                   // bf16 [512][128] (W_hh_enc)
#define WS_WATT  (WS_WHH  + 131072u)                  // bf16 [176][160] (W_att padded 168->176, 135->160)
#define WS_WIHD  (WS_WATT + 56320u)                   // bf16 [512][128] (W_ih_dec)
#define WS_WHHD  (WS_WIHD + 131072u)                  // bf16 [512][128] (W_hh_dec)

// LDS byte layout (total 286,464 B -> 1 workgroup / 320KB WGP)
#define L_PRE   0u         // f32 [32][512]  gate preacts / [32][176] attn logits
#define L_SBUF  65536u     // f32 [32][128]  enc h (f32) -> dec c_new state
#define L_CBUF  81920u     // f32 [32][128]  encoder cell state
#define L_HBF   98304u     // bf16[32][128]  recurrent state as WMMA A operand
#define L_CMB   106496u    // bf16[32][128]  attention combine
#define L_XYBF  114688u    // bf16[32][32]   x_s (enc) / y_prev padded (dec)
#define L_YBUF  116736u    // f32 [32][8]    decoder out (F=7)
#define L_BENC  117760u    // f32 [512]      b_ih_enc+b_hh_enc
#define L_BDEC  119808u    // f32 [512]      b_ih_dec+b_hh_dec
#define L_BATT  121856u    // f32 [176]      b_att padded with -1e30
#define L_WLDS  122624u    // bf16 weight stage: enc=[wihx 32K|whh 128K]; dec=whhd 128K
#define LDS_BYTES 286464u  // 122624 + 163840

// ---- CDNA5 async global->LDS bulk copy (ASYNCcnt-tracked) ------------------
// VDST = WG-relative LDS byte offset (VGPR), VADDR = 64-bit global address.
__device__ __forceinline__ void async_g2l_b128(unsigned lds_off, const void* gaddr) {
  asm volatile("global_load_async_to_lds_b128 %0, %1, off"
               :: "v"(lds_off), "v"((unsigned long long)(uintptr_t)gaddr)
               : "memory");
}
__device__ __forceinline__ void wait_async0() {
  asm volatile("s_wait_asynccnt 0" ::: "memory");
}

// ---- WMMA fragment helpers (16x16x32 bf16) --------------------------------
__device__ __forceinline__ v16bf load_frag(const bf16_t* base, int ld,
                                           int rbase, int kbase, int lane) {
  const int hi = lane >> 4;
  const int r  = lane & 15;
  const bf16_t* row = base + (size_t)(rbase + r) * ld + kbase + (hi ? 8 : 0);
  union { v16bf v; unsigned u[8]; } f;
#pragma unroll
  for (int j = 0; j < 4; ++j) f.u[j]     = *(const unsigned*)(row + 2 * j);
#pragma unroll
  for (int j = 0; j < 4; ++j) f.u[4 + j] = *(const unsigned*)(row + 16 + 2 * j);
  return f.v;
}

__device__ __forceinline__ void store_ctile(float* buf, int ld, int mbase,
                                            int nbase, v8f c, int lane) {
  const int hi = lane >> 4;
  const int n  = lane & 15;
  float* p = buf + (size_t)(mbase + (hi ? 8 : 0)) * ld + nbase + n;
#pragma unroll
  for (int r = 0; r < 8; ++r) p[(size_t)r * ld] = c[r];
}

__device__ __forceinline__ v8f bwmma(v16bf a, v16bf b, v8f c) {
  return __builtin_amdgcn_wmma_f32_16x16x32_bf16(false, a, false, b,
                                                 (short)0, c, false, false);
}

// ---- fast activations: raw v_rcp_f32 + v_exp_f32 (serial-path critical) ---
__device__ __forceinline__ float fast_rcp(float x) {
  return __builtin_amdgcn_rcpf(x);
}
__device__ __forceinline__ float sigf(float x) {
  return fast_rcp(1.0f + __expf(-x));
}
__device__ __forceinline__ float tanh_fast(float x) {
  // tanh(x) = 1 - 2/(1+exp(2x));  saturates correctly for |x| large
  return 1.0f - 2.0f * fast_rcp(1.0f + __expf(2.0f * x));
}

// ---- init: convert weights to padded bf16 layouts in d_ws -----------------
__global__ void convert_weights_kernel(const float* __restrict__ Wih_enc,   // [512][7]
                                       const float* __restrict__ Whh_enc,   // [512][128]
                                       const float* __restrict__ Watt,      // [168][135]
                                       const float* __restrict__ Wihd,      // [512][128]
                                       const float* __restrict__ Whhd,      // [512][128]
                                       bf16_t* __restrict__ wihx,
                                       bf16_t* __restrict__ whh,
                                       bf16_t* __restrict__ watt,
                                       bf16_t* __restrict__ wihd,
                                       bf16_t* __restrict__ whhd) {
  const int i = blockIdx.x * blockDim.x + threadIdx.x;   // 0 .. 65535
  if (i < 512 * 32) {
    int g = i >> 5, f = i & 31;
    wihx[i] = (bf16_t)((f < F_) ? Wih_enc[g * F_ + f] : 0.0f);
  }
  if (i < 176 * 160) {
    int n = i / 160, k = i % 160;
    watt[i] = (bf16_t)((n < S_ && k < (H_ + F_)) ? Watt[n * (H_ + F_) + k] : 0.0f);
  }
  if (i < 512 * 128) {
    whh[i]  = (bf16_t)Whh_enc[i];
    wihd[i] = (bf16_t)Wihd[i];
    whhd[i] = (bf16_t)Whhd[i];
  }
}

// ---- the fused persistent kernel ------------------------------------------
__global__ void fused_encdec_kernel(const float* __restrict__ xb,        // [B][S][F]
                                    const float* __restrict__ b_ih_enc,  // [512]
                                    const float* __restrict__ b_hh_enc,  // [512]
                                    const float* __restrict__ b_att,     // [168]
                                    const float* __restrict__ b_ih_dec,  // [512]
                                    const float* __restrict__ b_hh_dec,  // [512]
                                    const float* __restrict__ W_out,     // [7][128]
                                    const float* __restrict__ b_out,     // [7]
                                    const float* __restrict__ W_fin,     // [1][7]
                                    const float* __restrict__ b_fin,     // [1]
                                    const bf16_t* __restrict__ wenc_pack,// [wihx|whh] 160KB
                                    const bf16_t* __restrict__ watt,     // [176][160]
                                    const bf16_t* __restrict__ wihd,     // [512][128]
                                    const bf16_t* __restrict__ whhd,     // [512][128]
                                    bf16_t* __restrict__ enc,            // [B][S][H]
                                    float* __restrict__ out) {           // [B][P]
  extern __shared__ char smem[];
  float*  pre  = (float*)(smem + L_PRE);
  float*  sbuf = (float*)(smem + L_SBUF);
  float*  cbuf = (float*)(smem + L_CBUF);
  bf16_t* hbf  = (bf16_t*)(smem + L_HBF);
  bf16_t* cmb  = (bf16_t*)(smem + L_CMB);
  bf16_t* xybf = (bf16_t*)(smem + L_XYBF);
  float*  ybuf = (float*)(smem + L_YBUF);
  float*  benc = (float*)(smem + L_BENC);
  float*  bdec = (float*)(smem + L_BDEC);
  float*  batt = (float*)(smem + L_BATT);
  bf16_t* wihxl = (bf16_t*)(smem + L_WLDS);            // enc: [512][32]
  bf16_t* whhl  = (bf16_t*)(smem + L_WLDS + 32768u);   // enc: [512][128]
  bf16_t* whhdl = (bf16_t*)(smem + L_WLDS);            // dec: [512][128]

  const int tid  = threadIdx.x;
  const int lane = tid & 31;
  const int wave = tid >> 5;
  const int b0   = blockIdx.x * BT;

  // ---- kick off async weight staging: [wihx|whh] 160KB global -> LDS ----
  for (unsigned i = tid; i < (163840u >> 4); i += NT_THREADS)
    async_g2l_b128(L_WLDS + (i << 4), (const char*)wenc_pack + ((size_t)i << 4));

  // ---- init LDS state (overlaps with async copy) ----
  for (int i = tid; i < G_; i += NT_THREADS) {
    benc[i] = b_ih_enc[i] + b_hh_enc[i];
    bdec[i] = b_ih_dec[i] + b_hh_dec[i];
  }
  for (int i = tid; i < 176; i += NT_THREADS)
    batt[i] = (i < S_) ? b_att[i] : -1.0e30f;
  for (int i = tid; i < BT * H_; i += NT_THREADS) {
    cbuf[i] = 0.0f; sbuf[i] = 0.0f; hbf[i] = (bf16_t)0.0f;
  }
  for (int i = tid; i < BT * 32; i += NT_THREADS) xybf[i] = (bf16_t)0.0f;
  wait_async0();
  __syncthreads();

  // =================== encoder: 168 sequential LSTM steps ===================
  for (int s = 0; s < S_; ++s) {
    // stage x_s (bf16, cols 7..31 stay zero)
    for (int i = tid; i < BT * F_; i += NT_THREADS) {
      int b = i / F_, f = i % F_;
      xybf[b * 32 + f] = (bf16_t)xb[((size_t)(b0 + b) * S_ + s) * F_ + f];
    }
    __syncthreads();

    // pre[32x512] = x@W_ih^T + h@W_hh^T   (K=32 padded + K=128, B in LDS)
    {
      const int mt = wave & 1;
      v16bf ax  = load_frag(xybf, 32, mt * 16, 0, lane);
      v16bf ah0 = load_frag(hbf, H_, mt * 16,  0, lane);
      v16bf ah1 = load_frag(hbf, H_, mt * 16, 32, lane);
      v16bf ah2 = load_frag(hbf, H_, mt * 16, 64, lane);
      v16bf ah3 = load_frag(hbf, H_, mt * 16, 96, lane);
      for (int nt = (wave >> 1); nt < 32; nt += 4) {
        v8f acc = {};
        acc = bwmma(ax,  load_frag(wihxl, 32, nt * 16,  0, lane), acc);
        acc = bwmma(ah0, load_frag(whhl, H_, nt * 16,  0, lane), acc);
        acc = bwmma(ah1, load_frag(whhl, H_, nt * 16, 32, lane), acc);
        acc = bwmma(ah2, load_frag(whhl, H_, nt * 16, 64, lane), acc);
        acc = bwmma(ah3, load_frag(whhl, H_, nt * 16, 96, lane), acc);
        store_ctile(pre, G_, mt * 16, nt * 16, acc, lane);
      }
    }
    __syncthreads();

    // gates + state update, write h to encoder_output (bf16 -> L2)
    for (int i = tid; i < BT * H_; i += NT_THREADS) {
      int b = i >> 7, hh = i & 127;
      const float* pr = pre + (size_t)b * G_;
      float ig = sigf     (pr[hh]        + benc[hh]);
      float fg = sigf     (pr[128 + hh]  + benc[128 + hh]);
      float gg = tanh_fast(pr[256 + hh]  + benc[256 + hh]);
      float og = sigf     (pr[384 + hh]  + benc[384 + hh]);
      float c  = fg * cbuf[i] + ig * gg;
      float h  = og * tanh_fast(c);
      cbuf[i] = c;
      sbuf[i] = h;                      // f32 h (decoder initial state)
      hbf[i]  = (bf16_t)h;
      enc[((size_t)(b0 + b) * S_ + s) * H_ + hh] = (bf16_t)h;
    }
    __syncthreads();
  }

  // ---- decoder transition: stage whhd into LDS (encoder weights dead now),
  //      init y_prev = xb[:, S-1, :] ---------------------------------------
  for (unsigned i = tid; i < (131072u >> 4); i += NT_THREADS)
    async_g2l_b128(L_WLDS + (i << 4), (const char*)whhd + ((size_t)i << 4));
  for (int i = tid; i < BT * 32; i += NT_THREADS) xybf[i] = (bf16_t)0.0f;
  __syncthreads();
  for (int i = tid; i < BT * F_; i += NT_THREADS) {
    int b = i / F_, f = i % F_;
    xybf[b * 32 + f] = (bf16_t)xb[((size_t)(b0 + b) * S_ + (S_ - 1)) * F_ + f];
  }
  wait_async0();
  __syncthreads();

  // =================== decoder: 96 sequential steps =========================
  for (int p = 0; p < P_; ++p) {
    // attention logits [32x176] = [h|y]@W_att^T  (K = 128 + 32 padded)
    for (int t = wave; t < 22; t += 8) {
      int mt = t / 11, nt = t % 11;
      v8f acc = {};
      acc = bwmma(load_frag(hbf, H_, mt * 16,  0, lane), load_frag(watt, 160, nt * 16,   0, lane), acc);
      acc = bwmma(load_frag(hbf, H_, mt * 16, 32, lane), load_frag(watt, 160, nt * 16,  32, lane), acc);
      acc = bwmma(load_frag(hbf, H_, mt * 16, 64, lane), load_frag(watt, 160, nt * 16,  64, lane), acc);
      acc = bwmma(load_frag(hbf, H_, mt * 16, 96, lane), load_frag(watt, 160, nt * 16,  96, lane), acc);
      acc = bwmma(load_frag(xybf, 32, mt * 16, 0, lane), load_frag(watt, 160, nt * 16, 128, lane), acc);
      store_ctile(pre, 176, mt * 16, nt * 16, acc, lane);
    }
    __syncthreads();

    // softmax: 8 lanes per row, 4 rows per wave32, wave-local __shfl_xor
    {
      const int row = tid >> 3;          // 32 rows
      const int q   = tid & 7;           // 8 lanes per row (same wave)
      float* rp = pre + (size_t)row * 176;
      float mx = -3.0e38f;
      for (int k = q; k < 176; k += 8) {
        float v = rp[k] + batt[k];
        rp[k] = v;
        mx = fmaxf(mx, v);
      }
#pragma unroll
      for (int off = 1; off < 8; off <<= 1)
        mx = fmaxf(mx, __shfl_xor(mx, off, 32));
      float sum = 0.0f;
      for (int k = q; k < 176; k += 8) {
        float e = __expf(rp[k] - mx);
        rp[k] = e;
        sum += e;
      }
#pragma unroll
      for (int off = 1; off < 8; off <<= 1)
        sum += __shfl_xor(sum, off, 32);
      float inv = fast_rcp(sum);
      for (int k = q; k < 176; k += 8) rp[k] *= inv;
    }
    __syncthreads();

    // combine[b][h] = sum_s w[b][s] * enc[b][s][h]  (L2-resident bf16 reads)
    for (int i = tid; i < BT * H_; i += NT_THREADS) {
      int b = i >> 7, hh = i & 127;
      const bf16_t* e  = enc + (size_t)(b0 + b) * S_ * H_ + hh;
      const float*  wr = pre + (size_t)b * 176;
      float acc = 0.0f;
      for (int k = 0; k < S_; ++k) {
        if ((k & 15) == 0 && k + 32 < S_)
          __builtin_prefetch(e + (size_t)(k + 32) * H_, 0, 1);
        acc += wr[k] * (float)e[(size_t)k * H_];
      }
      cmb[i] = (bf16_t)acc;
    }
    __syncthreads();

    // gate preacts [32x512] = combine@W_ih_dec^T (L2) + h@W_hh_dec^T (LDS)
    {
      const int mt = wave & 1;
      v16bf ac0 = load_frag(cmb, H_, mt * 16,  0, lane);
      v16bf ac1 = load_frag(cmb, H_, mt * 16, 32, lane);
      v16bf ac2 = load_frag(cmb, H_, mt * 16, 64, lane);
      v16bf ac3 = load_frag(cmb, H_, mt * 16, 96, lane);
      v16bf ah0 = load_frag(hbf, H_, mt * 16,  0, lane);
      v16bf ah1 = load_frag(hbf, H_, mt * 16, 32, lane);
      v16bf ah2 = load_frag(hbf, H_, mt * 16, 64, lane);
      v16bf ah3 = load_frag(hbf, H_, mt * 16, 96, lane);
      for (int nt = (wave >> 1); nt < 32; nt += 4) {
        v8f acc = {};
        acc = bwmma(ac0, load_frag(wihd,  H_, nt * 16,  0, lane), acc);
        acc = bwmma(ac1, load_frag(wihd,  H_, nt * 16, 32, lane), acc);
        acc = bwmma(ac2, load_frag(wihd,  H_, nt * 16, 64, lane), acc);
        acc = bwmma(ac3, load_frag(wihd,  H_, nt * 16, 96, lane), acc);
        acc = bwmma(ah0, load_frag(whhdl, H_, nt * 16,  0, lane), acc);
        acc = bwmma(ah1, load_frag(whhdl, H_, nt * 16, 32, lane), acc);
        acc = bwmma(ah2, load_frag(whhdl, H_, nt * 16, 64, lane), acc);
        acc = bwmma(ah3, load_frag(whhdl, H_, nt * 16, 96, lane), acc);
        store_ctile(pre, G_, mt * 16, nt * 16, acc, lane);
      }
    }
    __syncthreads();

    // c_new = f*prev + i*g   (o gate unused by reference decoder)
    for (int i = tid; i < BT * H_; i += NT_THREADS) {
      int b = i >> 7, hh = i & 127;
      const float* pr = pre + (size_t)b * G_;
      float ig = sigf     (pr[hh]       + bdec[hh]);
      float fg = sigf     (pr[128 + hh] + bdec[128 + hh]);
      float gg = tanh_fast(pr[256 + hh] + bdec[256 + hh]);
      float cn = fg * sbuf[i] + ig * gg;
      sbuf[i] = cn;
      hbf[i]  = (bf16_t)cn;
    }
    __syncthreads();

    // out[b][f] = c_new @ W_out^T + b_out  (32x7 x K=128, VALU)
    if (tid < BT * F_) {
      int b = tid / F_, f = tid % F_;
      const float* st = sbuf + (size_t)b * H_;
      const float* wr = W_out + (size_t)f * H_;
      float o = b_out[f];
      for (int k = 0; k < H_; ++k) o += st[k] * wr[k];
      ybuf[b * 8 + f]  = o;
      xybf[b * 32 + f] = (bf16_t)o;   // y_prev for next step's attention
    }
    __syncthreads();

    // final scalar: out @ W_fin^T + b_fin -> d_out[b][p]
    if (tid < BT) {
      float o = b_fin[0];
      for (int f = 0; f < F_; ++f) o += ybuf[tid * 8 + f] * W_fin[f];
      out[(size_t)(b0 + tid) * P_ + p] = o;
    }
    __syncthreads();
  }
}

// ---------------------------------------------------------------------------
extern "C" void kernel_launch(void* const* d_in, const int* in_sizes, int n_in,
                              void* d_out, int out_size, void* d_ws, size_t ws_size,
                              hipStream_t stream) {
  const float* xb       = (const float*)d_in[0];
  const float* W_ih_enc = (const float*)d_in[1];
  const float* W_hh_enc = (const float*)d_in[2];
  const float* b_ih_enc = (const float*)d_in[3];
  const float* b_hh_enc = (const float*)d_in[4];
  const float* W_att    = (const float*)d_in[5];
  const float* b_att    = (const float*)d_in[6];
  const float* W_ih_dec = (const float*)d_in[7];
  const float* W_hh_dec = (const float*)d_in[8];
  const float* b_ih_dec = (const float*)d_in[9];
  const float* b_hh_dec = (const float*)d_in[10];
  const float* W_out    = (const float*)d_in[11];
  const float* b_out    = (const float*)d_in[12];
  const float* W_fin    = (const float*)d_in[13];
  const float* b_fin    = (const float*)d_in[14];

  char* ws = (char*)d_ws;
  bf16_t* enc  = (bf16_t*)(ws + WS_ENC);
  bf16_t* wihx = (bf16_t*)(ws + WS_WIHX);   // start of packed [wihx|whh]
  bf16_t* whh  = (bf16_t*)(ws + WS_WHH);
  bf16_t* watt = (bf16_t*)(ws + WS_WATT);
  bf16_t* wihd = (bf16_t*)(ws + WS_WIHD);
  bf16_t* whhd = (bf16_t*)(ws + WS_WHHD);

  convert_weights_kernel<<<256, 256, 0, stream>>>(
      W_ih_enc, W_hh_enc, W_att, W_ih_dec, W_hh_dec,
      wihx, whh, watt, wihd, whhd);

  (void)hipFuncSetAttribute((const void*)fused_encdec_kernel,
                            hipFuncAttributeMaxDynamicSharedMemorySize,
                            (int)LDS_BYTES);

  fused_encdec_kernel<<<B_ / BT, NT_THREADS, LDS_BYTES, stream>>>(
      xb, b_ih_enc, b_hh_enc, b_att, b_ih_dec, b_hh_dec,
      W_out, b_out, W_fin, b_fin,
      /*wenc_pack=*/wihx, watt, wihd, whhd,
      enc, (float*)d_out);
}
